// DTGCN_66949950210310
// MI455X (gfx1250) — compile-verified, matrix operand order checked by reading, not run on hardware
//
#include <hip/hip_runtime.h>
#include <hip/hip_bf16.h>
#include <math.h>

typedef __attribute__((ext_vector_type(16))) _Float16 v16h;
typedef __attribute__((ext_vector_type(8)))  _Float16 v8h;
typedef __attribute__((ext_vector_type(8)))  float    v8f;

// Problem constants
#define NN   256
#define BATCH 32
#define TT   256
#define WWIN 32
#define HDIM 128
#define EMBD 64
#define NEDGE 8192
#define NSTEP 8
#define NCLS 16

// ---------------- workspace layout (bytes) ----------------
// shared region
static constexpr size_t OFF_SDENSE = 0;                       // 256*256 f32
static constexpr size_t OFF_DI     = OFF_SDENSE + 262144;     // 256 f32
static constexpr size_t OFF_ST     = OFF_DI + 1024;           // S^T f16 256*256 (row-major A)
static constexpr size_t OFF_WG     = OFF_ST + 131072;         // WgT  [64][160] f16
static constexpr size_t OFF_W1     = OFF_WG + 20480;          // W1T  [64][64]  f16
static constexpr size_t OFF_W2     = OFF_W1 + 8192;           // W2T
static constexpr size_t OFF_WZ     = OFF_W2 + 8192;           // WzT  [128][32] f16
static constexpr size_t OFF_WR     = OFF_WZ + 8192;
static constexpr size_t OFF_WH     = OFF_WR + 8192;
static constexpr size_t OFF_WLZ    = OFF_WH + 8192;           // WlzT [128][256] f16
static constexpr size_t OFF_WLR    = OFF_WLZ + 65536;
static constexpr size_t OFF_WLH    = OFF_WLR + 65536;
// per-batch region
static constexpr size_t OFF_BATCH  = 1u << 20;
static constexpr size_t WS_PB      = 2u << 20;   // stride per batch
static constexpr size_t PB_H32  = 0;             // 256*128 f32
static constexpr size_t PB_H16  = 131072;        // 256*128 f16
static constexpr size_t PB_VT   = 196608;        // 256*32 f16
static constexpr size_t PB_IT   = 212992;        // 256*160 f16
static constexpr size_t PB_TMPG = 294912;        // tmpGT [64][256] f16
static constexpr size_t PB_DF   = 327680;        // 256*64 f16
static constexpr size_t PB_DE1  = 360448;
static constexpr size_t PB_DE2  = 393216;
static constexpr size_t PB_M    = 425984;        // MT [256][256] f32 (transposed)
static constexpr size_t PB_HIST = 688128;        // 3 * 256*256 f16
static constexpr size_t PB_ANT  = 1081344;       // 256*256 f16 (transposed An, row-major A)
static constexpr size_t PB_VTW  = 1212416;       // vtWT [128][256] f16
static constexpr size_t PB_CATA = 1277952;       // 256*256 f16
static constexpr size_t PB_CATB = 1409024;       // 256*256 f16
static constexpr size_t PB_Z    = 1540096;       // 256*128 f32

// ---------------- WMMA fragment helpers (wave32, 16x16x32 f16) ----------------
// Fragment K-striping (16-bit A/B, ISA 7.12.2): lane half (l>=16) owns K-base +8;
// halves [0..7] at +0, halves [8..15] at +16. Both runs are 16B-aligned here.
__device__ __forceinline__ v16h frag_ld(const _Float16* p) {
  v8h lo = *(const v8h*)p;
  v8h hi = *(const v8h*)(p + 16);
  v16h f;
#pragma unroll
  for (int i = 0; i < 8; ++i) { f[i] = lo[i]; f[8 + i] = hi[i]; }
  return f;
}

// A 16x32 f16, row-major [M x K]. Lane l: row = l&15.
__device__ __forceinline__ v16h frag_a_rm(const _Float16* A, int lda, int row0, int k0) {
  const int l = threadIdx.x & 31;
  return frag_ld(A + (size_t)(row0 + (l & 15)) * lda + k0 + (((l >> 4) & 1) << 3));
}

// B 32x16 f16 given TRANSPOSED in memory: Bt[N x K] row-major, logical B[k][n] = Bt[n][k].
__device__ __forceinline__ v16h frag_b_tr(const _Float16* Bt, int ldt, int k0, int col0) {
  const int l = threadIdx.x & 31;
  return frag_ld(Bt + (size_t)(col0 + (l & 15)) * ldt + k0 + (((l >> 4) & 1) << 3));
}

__device__ __forceinline__ v8f mm(const _Float16* A, int lda, const _Float16* Bt, int ldt,
                                  int K, int row0, int col0) {
  v8f acc = {};
  for (int k0 = 0; k0 < K; k0 += 32)
    acc = __builtin_amdgcn_wmma_f32_16x16x32_f16(false, frag_a_rm(A, lda, row0, k0),
                                                 false, frag_b_tr(Bt, ldt, k0, col0),
                                                 (short)0, acc, false, false);
  return acc;
}

// C/D 16x16 f32 layout: lane l -> column l&15; VGPR j -> row j + 8*(l>=16).
// Transposed f16 store: per lane 8 contiguous halves -> single b128 store.
__device__ __forceinline__ void store_f16_tr(_Float16* Ct, int ldt, int row0, int col0, v8f a) {
  const int l = threadIdx.x & 31;
  v8h v;
#pragma unroll
  for (int j = 0; j < 8; ++j) v[j] = (_Float16)a[j];
  *(v8h*)(Ct + (size_t)(col0 + (l & 15)) * ldt + row0 + (((l >> 4) & 1) << 3)) = v;
}

// Transposed f32 store: per lane 8 contiguous floats (32B aligned).
__device__ __forceinline__ void store_f32_tr(float* Ct, int ldt, int row0, int col0, v8f a) {
  const int l = threadIdx.x & 31;
  *(v8f*)(Ct + (size_t)(col0 + (l & 15)) * ldt + row0 + (((l >> 4) & 1) << 3)) = a;
}

__device__ __forceinline__ float sigm(float v) { return 1.0f / (1.0f + expf(-v)); }

// ---------------- prep kernels ----------------
__global__ void k_zero_f32(float* p, int n) {
  int i = blockIdx.x * blockDim.x + threadIdx.x;
  if (i < n) p[i] = 0.0f;
}

__global__ void k_scatter(const int* ei, const float* ew, float* A, int E) {
  int e = blockIdx.x * blockDim.x + threadIdx.x;
  if (e < E) atomicAdd(&A[(size_t)ei[e] * NN + ei[E + e]], ew[e]);
}

__global__ void k_colnorm(const float* A, float* di) {
  int c = threadIdx.x;  // 256 threads, one column each
  float s = 1.0f;       // + self-loop from identity
  for (int r = 0; r < NN; ++r) s += A[(size_t)r * NN + c];
  di[c] = rsqrtf(s);    // s >= 1 always
}

// St[c][r] = di[r] * (A[r][c] + I) * di[c]   (i.e. S^T in f16, row-major A operand)
__global__ void k_buildSt(const float* A, const float* di, _Float16* St) {
  int i = blockIdx.x * blockDim.x + threadIdx.x;
  if (i < NN * NN) {
    int r = i >> 8, c = i & 255;
    float a = A[i] + (r == c ? 1.0f : 0.0f);
    St[(size_t)c * NN + r] = (_Float16)(di[r] * a * di[c]);
  }
}

// Transposing f32 -> f16 convert: s is [rows][cols] row-major, d becomes [cols][rows].
__global__ void k_cvt_tr(const float* s, _Float16* d, int rows, int cols) {
  int i = blockIdx.x * blockDim.x + threadIdx.x;
  if (i < rows * cols) {
    int r = i / cols, c = i % cols;
    d[(size_t)c * rows + r] = (_Float16)s[i];
  }
}

// ---------------- persistent per-batch kernel: full 8-step scan ----------------
__global__ void __launch_bounds__(256) dtgcn_main(
    const float* __restrict__ x,
    const float* __restrict__ bg, const float* __restrict__ b1, const float* __restrict__ b2,
    const float* __restrict__ bz, const float* __restrict__ br, const float* __restrict__ bh,
    const float* __restrict__ blz, const float* __restrict__ blr, const float* __restrict__ blh,
    const float* __restrict__ Wc, const float* __restrict__ bc,
    char* __restrict__ ws, float* __restrict__ out) {
  const int b   = blockIdx.x;
  const int tid = threadIdx.x;
  const int wv  = tid >> 5;
  const int lc  = tid & 15;                 // lane column inside 16x16 tile
  const int lr  = ((tid >> 4) & 1) << 3;    // row base (0 or 8)

  const _Float16* St    = (const _Float16*)(ws + OFF_ST);
  const _Float16* WgT   = (const _Float16*)(ws + OFF_WG);
  const _Float16* W1T   = (const _Float16*)(ws + OFF_W1);
  const _Float16* W2T   = (const _Float16*)(ws + OFF_W2);
  const _Float16* WzT   = (const _Float16*)(ws + OFF_WZ);
  const _Float16* WrT   = (const _Float16*)(ws + OFF_WR);
  const _Float16* WhT   = (const _Float16*)(ws + OFF_WH);
  const _Float16* WlzT  = (const _Float16*)(ws + OFF_WLZ);
  const _Float16* WlrT  = (const _Float16*)(ws + OFF_WLR);
  const _Float16* WlhT  = (const _Float16*)(ws + OFF_WLH);

  char* wb = ws + OFF_BATCH + (size_t)b * WS_PB;
  float*     h32   = (float*)(wb + PB_H32);
  _Float16*  h16   = (_Float16*)(wb + PB_H16);
  _Float16*  vt16  = (_Float16*)(wb + PB_VT);
  _Float16*  It16  = (_Float16*)(wb + PB_IT);
  _Float16*  tmpGT = (_Float16*)(wb + PB_TMPG);   // [64][256]
  _Float16*  df16  = (_Float16*)(wb + PB_DF);     // [256][64]
  _Float16*  de1   = (_Float16*)(wb + PB_DE1);    // [256][64]
  _Float16*  de2   = (_Float16*)(wb + PB_DE2);    // [256][64] (natural Bt layout)
  float*     MT    = (float*)(wb + PB_M);         // logical M[r][c] = MT[c*256+r]
  _Float16*  hist  = (_Float16*)(wb + PB_HIST);
  _Float16*  AnT   = (_Float16*)(wb + PB_ANT);    // [256][256] row-major A operand
  _Float16*  vtWT  = (_Float16*)(wb + PB_VTW);    // [128][256]
  _Float16*  catA  = (_Float16*)(wb + PB_CATA);   // [256][256]
  _Float16*  catB  = (_Float16*)(wb + PB_CATB);
  float*     Zbuf  = (float*)(wb + PB_Z);

  __shared__ float s_q[NN];
  __shared__ float s_red[NCLS];

  // h0 = 0
  for (int i = tid; i < NN * HDIM; i += 256) { h32[i] = 0.0f; h16[i] = (_Float16)0.0f; }
  __syncthreads();

  const float* xb = x + (size_t)b * NN * TT;

  for (int t = 0; t < NSTEP; ++t) {
    // ---- Phase A: vt (f16), It = [vt | h] (f16), cat second halves = h ----
    for (int i = tid; i < NN * WWIN; i += 256) {
      int n = i >> 5, w = i & 31;
      vt16[i] = (_Float16)xb[n * TT + t * WWIN + w];
    }
    for (int i = tid; i < NN * (WWIN + HDIM); i += 256) {
      int n = i / 160, c = i % 160;
      float v = (c < WWIN) ? xb[n * TT + t * WWIN + c] : (float)h16[n * HDIM + (c - WWIN)];
      It16[i] = (_Float16)v;
    }
    for (int i = tid; i < NN * HDIM; i += 256) {
      int n = i >> 7, j = i & 127;
      _Float16 hv = h16[i];
      catA[(size_t)n * 256 + 128 + j] = hv;
      catB[(size_t)n * 256 + 128 + j] = hv;
    }
    __syncthreads();

    // ---- Phase B: tmpG = It @ Wg  [256x160][160x64], stored transposed ----
    for (int tile = wv; tile < 64; tile += 8) {
      int r0 = (tile >> 2) << 4, c0 = (tile & 3) << 4;
      store_f16_tr(tmpGT, 256, r0, c0, mm(It16, 160, WgT, 160, 160, r0, c0));
    }
    __syncthreads();

    // ---- Phase C: df = S^T @ tmpG + bg  [256x256][256x64] ----
    for (int tile = wv; tile < 64; tile += 8) {
      int r0 = (tile >> 2) << 4, c0 = (tile & 3) << 4;
      v8f acc = mm(St, 256, tmpGT, 256, 256, r0, c0);
      float bb = bg[c0 + lc];
#pragma unroll
      for (int j = 0; j < 8; ++j)
        df16[(size_t)(r0 + lr + j) * 64 + c0 + lc] = (_Float16)(acc[j] + bb);
    }
    __syncthreads();

    // ---- Phase D: de1 = tanh(df@W1+b1), de2 = tanh(df@W2+b2)  [256x64][64x64] ----
    for (int tile = wv; tile < 64; tile += 8) {
      int r0 = (tile >> 2) << 4, c0 = (tile & 3) << 4;
      v8f a1 = mm(df16, 64, W1T, 64, 64, r0, c0);
      v8f a2 = mm(df16, 64, W2T, 64, 64, r0, c0);
      float bb1 = b1[c0 + lc], bb2 = b2[c0 + lc];
#pragma unroll
      for (int j = 0; j < 8; ++j) {
        size_t idx = (size_t)(r0 + lr + j) * 64 + c0 + lc;
        de1[idx] = (_Float16)tanhf(a1[j] + bb1);
        de2[idx] = (_Float16)tanhf(a2[j] + bb2);
      }
    }
    __syncthreads();

    // ---- Phase E: M = de1 @ de2^T  [256x64][64x256], stored transposed (v8f) ----
    for (int tile = wv; tile < 256; tile += 8) {
      int r0 = (tile >> 4) << 4, c0 = (tile & 15) << 4;
      store_f32_tr(MT, 256, r0, c0, mm(de1, 64, de2, 64, 64, r0, c0));
    }
    __syncthreads();

    // ---- Phase F: Et = relu(tanh(M - M^T)) -> hist[t%3]  (M[r][c] = MT[c*256+r]) ----
    _Float16* histT = hist + (size_t)(t % 3) * NN * NN;
    for (int i = tid; i < NN * NN; i += 256) {
      int r = i >> 8, c = i & 255;
      float e = tanhf(MT[(size_t)c * 256 + r] - MT[(size_t)r * 256 + c]);
      histT[i] = (_Float16)(e > 0.0f ? e : 0.0f);
    }
    __syncthreads();

    // ---- Phase G: Mt avg, threshold, +I, symmetric normalize -> AnT (f16) ----
    s_q[tid] = 0.0f;
    __syncthreads();
    {
      const int cnt = (t < 2) ? (t + 1) : 3;
      const float inv = 1.0f / (float)cnt;
      for (int i = tid; i < NN * NN; i += 256) {
        int r = i >> 8, c = i & 255;
        float s = 0.0f;
        for (int j = 0; j < cnt; ++j) s += (float)hist[(size_t)((t - j + 3) % 3) * NN * NN + i];
        s *= inv;
        float v = (s > 1e-8f) ? s : 0.0f;
        float ah = v + (r == c ? 1.0f : 0.0f);
        MT[i] = ah;                   // reuse M buffer as Ah scratch (logical [r][c] here)
        atomicAdd(&s_q[c], ah);       // column sums (LDS ds_add_f32)
      }
    }
    __syncthreads();
    { float d = s_q[tid]; s_q[tid] = (d > 0.0f) ? rsqrtf(d) : 0.0f; }
    __syncthreads();
    for (int i = tid; i < NN * NN; i += 256) {
      int r = i >> 8, c = i & 255;
      AnT[(size_t)c * 256 + r] = (_Float16)(s_q[r] * MT[i] * s_q[c]);
    }
    __syncthreads();

    // ---- Z gate: vtWT = (vt@Wz)^T ; catA[:, :128] = AnT@vtW + bz ----
    for (int tile = wv; tile < 128; tile += 8) {
      int r0 = (tile >> 3) << 4, c0 = (tile & 7) << 4;
      store_f16_tr(vtWT, 256, r0, c0, mm(vt16, 32, WzT, 32, 32, r0, c0));
    }
    __syncthreads();
    for (int tile = wv; tile < 128; tile += 8) {
      int r0 = (tile >> 3) << 4, c0 = (tile & 7) << 4;
      v8f acc = mm(AnT, 256, vtWT, 256, 256, r0, c0);
      float bb = bz[c0 + lc];
#pragma unroll
      for (int j = 0; j < 8; ++j)
        catA[(size_t)(r0 + lr + j) * 256 + c0 + lc] = (_Float16)(acc[j] + bb);
    }
    __syncthreads();

    // ---- R gate conv: vtWT = (vt@Wr)^T ; catB[:, :128] = AnT@vtW + br ----
    for (int tile = wv; tile < 128; tile += 8) {
      int r0 = (tile >> 3) << 4, c0 = (tile & 7) << 4;
      store_f16_tr(vtWT, 256, r0, c0, mm(vt16, 32, WrT, 32, 32, r0, c0));
    }
    __syncthreads();
    for (int tile = wv; tile < 128; tile += 8) {
      int r0 = (tile >> 3) << 4, c0 = (tile & 7) << 4;
      v8f acc = mm(AnT, 256, vtWT, 256, 256, r0, c0);
      float bb = br[c0 + lc];
#pragma unroll
      for (int j = 0; j < 8; ++j)
        catB[(size_t)(r0 + lr + j) * 256 + c0 + lc] = (_Float16)(acc[j] + bb);
    }
    __syncthreads();

    // ---- Z = sigmoid(catA @ Wlz + blz) -> Zbuf f32 ----
    for (int tile = wv; tile < 128; tile += 8) {
      int r0 = (tile >> 3) << 4, c0 = (tile & 7) << 4;
      v8f acc = mm(catA, 256, WlzT, 256, 256, r0, c0);
      float bb = blz[c0 + lc];
#pragma unroll
      for (int j = 0; j < 8; ++j)
        Zbuf[(size_t)(r0 + lr + j) * 128 + c0 + lc] = sigm(acc[j] + bb);
    }
    __syncthreads();

    // ---- R = sigmoid(catB @ Wlr + blr); catA[:, 128:] = h * R (f16) ----
    for (int tile = wv; tile < 128; tile += 8) {
      int r0 = (tile >> 3) << 4, c0 = (tile & 7) << 4;
      v8f acc = mm(catB, 256, WlrT, 256, 256, r0, c0);
      float bb = blr[c0 + lc];
#pragma unroll
      for (int j = 0; j < 8; ++j) {
        int row = r0 + lr + j, col = c0 + lc;
        float rr = sigm(acc[j] + bb);
        catA[(size_t)row * 256 + 128 + col] = (_Float16)(h32[(size_t)row * 128 + col] * rr);
      }
    }
    __syncthreads();

    // ---- H gate conv: vtWT = (vt@Wh)^T ; catA[:, :128] = AnT@vtW + bh ----
    for (int tile = wv; tile < 128; tile += 8) {
      int r0 = (tile >> 3) << 4, c0 = (tile & 7) << 4;
      store_f16_tr(vtWT, 256, r0, c0, mm(vt16, 32, WhT, 32, 32, r0, c0));
    }
    __syncthreads();
    for (int tile = wv; tile < 128; tile += 8) {
      int r0 = (tile >> 3) << 4, c0 = (tile & 7) << 4;
      v8f acc = mm(AnT, 256, vtWT, 256, 256, r0, c0);
      float bb = bh[c0 + lc];
#pragma unroll
      for (int j = 0; j < 8; ++j)
        catA[(size_t)(r0 + lr + j) * 256 + c0 + lc] = (_Float16)(acc[j] + bb);
    }
    __syncthreads();

    // ---- Ht = tanh(catA @ Wlh + blh); h = Z*h + (1-Z)*Ht ----
    for (int tile = wv; tile < 128; tile += 8) {
      int r0 = (tile >> 3) << 4, c0 = (tile & 7) << 4;
      v8f acc = mm(catA, 256, WlhT, 256, 256, r0, c0);
      float bb = blh[c0 + lc];
#pragma unroll
      for (int j = 0; j < 8; ++j) {
        size_t idx = (size_t)(r0 + lr + j) * 128 + c0 + lc;
        float ht = tanhf(acc[j] + bb);
        float z  = Zbuf[idx];
        float hn = z * h32[idx] + (1.0f - z) * ht;
        h32[idx] = hn;
        h16[idx] = (_Float16)hn;
      }
    }
    __syncthreads();
  }

  // ---- classifier: out[b] = h.flatten() @ Wc + bc ----
  if (tid < NCLS) s_red[tid] = 0.0f;
  __syncthreads();
  float loc[NCLS];
#pragma unroll
  for (int c = 0; c < NCLS; ++c) loc[c] = 0.0f;
  for (int i = tid; i < NN * HDIM; i += 256) {
    float hv = h32[i];
    const float* wrow = Wc + (size_t)i * NCLS;
#pragma unroll
    for (int c = 0; c < NCLS; ++c) loc[c] += hv * wrow[c];
  }
#pragma unroll
  for (int c = 0; c < NCLS; ++c) atomicAdd(&s_red[c], loc[c]);
  __syncthreads();
  if (tid < NCLS) out[(size_t)b * NCLS + tid] = s_red[tid] + bc[tid];
}

// ---------------- host launch ----------------
extern "C" void kernel_launch(void* const* d_in, const int* in_sizes, int n_in,
                              void* d_out, int out_size, void* d_ws, size_t ws_size,
                              hipStream_t stream) {
  (void)in_sizes; (void)n_in; (void)out_size; (void)ws_size;
  const float* x   = (const float*)d_in[0];
  const int*   ei  = (const int*)d_in[1];
  const float* ew  = (const float*)d_in[2];
  const float* Wg  = (const float*)d_in[3];  const float* bg  = (const float*)d_in[4];
  const float* W1  = (const float*)d_in[5];  const float* b1  = (const float*)d_in[6];
  const float* W2  = (const float*)d_in[7];  const float* b2  = (const float*)d_in[8];
  const float* Wz  = (const float*)d_in[9];  const float* bz  = (const float*)d_in[10];
  const float* Wr  = (const float*)d_in[11]; const float* br  = (const float*)d_in[12];
  const float* Wh  = (const float*)d_in[13]; const float* bh  = (const float*)d_in[14];
  const float* Wlz = (const float*)d_in[15]; const float* blz = (const float*)d_in[16];
  const float* Wlr = (const float*)d_in[17]; const float* blr = (const float*)d_in[18];
  const float* Wlh = (const float*)d_in[19]; const float* blh = (const float*)d_in[20];
  const float* Wc  = (const float*)d_in[21]; const float* bc  = (const float*)d_in[22];

  char* ws = (char*)d_ws;
  float*    Sd  = (float*)(ws + OFF_SDENSE);
  float*    di  = (float*)(ws + OFF_DI);
  _Float16* St  = (_Float16*)(ws + OFF_ST);

  k_zero_f32<<<256, 256, 0, stream>>>(Sd, NN * NN);
  k_scatter<<<32, 256, 0, stream>>>(ei, ew, Sd, NEDGE);
  k_colnorm<<<1, 256, 0, stream>>>(Sd, di);
  k_buildSt<<<256, 256, 0, stream>>>(Sd, di, St);

  // Transposing f32->f16 weight conversions (B operands stored as Bt[N][K]).
  k_cvt_tr<<<40, 256, 0, stream>>>(Wg,  (_Float16*)(ws + OFF_WG),  160, 64);
  k_cvt_tr<<<16, 256, 0, stream>>>(W1,  (_Float16*)(ws + OFF_W1),  64, 64);
  k_cvt_tr<<<16, 256, 0, stream>>>(W2,  (_Float16*)(ws + OFF_W2),  64, 64);
  k_cvt_tr<<<16, 256, 0, stream>>>(Wz,  (_Float16*)(ws + OFF_WZ),  32, 128);
  k_cvt_tr<<<16, 256, 0, stream>>>(Wr,  (_Float16*)(ws + OFF_WR),  32, 128);
  k_cvt_tr<<<16, 256, 0, stream>>>(Wh,  (_Float16*)(ws + OFF_WH),  32, 128);
  k_cvt_tr<<<128, 256, 0, stream>>>(Wlz, (_Float16*)(ws + OFF_WLZ), 256, 128);
  k_cvt_tr<<<128, 256, 0, stream>>>(Wlr, (_Float16*)(ws + OFF_WLR), 256, 128);
  k_cvt_tr<<<128, 256, 0, stream>>>(Wlh, (_Float16*)(ws + OFF_WLH), 256, 128);

  dtgcn_main<<<BATCH, 256, 0, stream>>>(x, bg, b1, b2, bz, br, bh, blz, blr, blh,
                                        Wc, bc, ws, (float*)d_out);
}
